// BertLayerForDecoder_55009941127229
// MI455X (gfx1250) — matrix operand order repaired and checked by reading, not compiled, not run on hardware
//
#include <hip/hip_runtime.h>
#include <hip/hip_bf16.h>

typedef __bf16 bf16;
typedef __attribute__((ext_vector_type(16))) __bf16 v16bf;
typedef __attribute__((ext_vector_type(8)))  float  v8f;

#define S_LEN   2048
#define D_MODEL 1024
#define N_HEAD  16
#define D_HEAD  64
#define D_FF    4096
#define BATCH   2
#define M_ROWS  (BATCH * S_LEN)   // 4096

// ---------------------------------------------------------------------------
// CDNA5 async tensor path: global -> LDS direct copy, 16B per lane, tracked
// by ASYNCcnt (ISA 10.4 / 15.18.3 op 98).  Generic LDS pointers on AMDGPU
// carry the LDS byte offset in their low 32 bits (ISA 10.2 aperture rules),
// so truncation yields the VDST LDS address the instruction expects.
// ---------------------------------------------------------------------------
__device__ inline void async_copy_b128(void* lds_dst, const void* gsrc) {
  unsigned loff = (unsigned)(unsigned long long)lds_dst;
  asm volatile("global_load_async_to_lds_b128 %0, %1, off"
               :: "v"(loff), "v"(gsrc)
               : "memory");
}
__device__ inline void async_fence() {        // drain this wave's ASYNCcnt
  asm volatile("s_wait_asynccnt 0x0" ::: "memory");
}

// ---------------------------------------------------------------------------
// WMMA helpers (CDNA5 gfx1250, wave32).  Layouts per ISA 7.12.2:
//  A (16x32 bf16): lane 0-15 -> M=lane, K pairs {0..7,16..23}; lanes 16-31 +8.
//  B (32x16 bf16): lane 0-15 -> N=lane, K pairs {0..7,16..23}; lanes 16-31 +8.
//  C/D (16x16 f32): lane -> N=lane&15, VGPR r -> M = r + 8*(lane>=16).
// ---------------------------------------------------------------------------
__device__ inline v8f wmma_bf16(v16bf a, v16bf b, v8f c) {
  return __builtin_amdgcn_wmma_f32_16x16x32_bf16(false, a, false, b, (short)0, c,
                                                 false, false);
}

// A-fragment from row-major [M x K] LDS tile (stride in elements, 4B-aligned rows)
__device__ inline v16bf frag_a(const bf16* __restrict__ p, int stride, int lane) {
  const int hh = lane >> 4, m = lane & 15;
  const bf16* row = p + m * stride;
  union { unsigned int u[8]; v16bf h; } cv;
#pragma unroll
  for (int v = 0; v < 8; ++v) {
    int k = ((v & 3) << 1) + ((v >> 2) << 4) + (hh << 3);
    cv.u[v] = *(const unsigned int*)(row + k);
  }
  return cv.h;
}

// B-fragment from row-major [K x N] LDS tile
__device__ inline v16bf frag_b(const bf16* __restrict__ p, int stride, int lane) {
  const int hh = lane >> 4, n = lane & 15;
  v16bf b;
#pragma unroll
  for (int s = 0; s < 16; ++s) {
    int k = (s & 7) + ((s >> 3) << 4) + (hh << 3);
    b[s] = p[k * stride + n];
  }
  return b;
}

// B-fragment with on-the-fly transpose: logical B[k][n] = tile[n][k]
// (tile stored row-major [N x K]); used for Q*K^T without a transposed stage.
__device__ inline v16bf frag_b_T(const bf16* __restrict__ p, int stride, int lane) {
  const int hh = lane >> 4, n = lane & 15;
  const bf16* row = p + n * stride;
  union { unsigned int u[8]; v16bf h; } cv;
#pragma unroll
  for (int v = 0; v < 8; ++v) {
    int k = ((v & 3) << 1) + ((v >> 2) << 4) + (hh << 3);
    cv.u[v] = *(const unsigned int*)(row + k);
  }
  return cv.h;
}

// ---------------------------------------------------------------------------
// fp32 -> bf16 conversion
// ---------------------------------------------------------------------------
__global__ void cvt_kernel(const float* __restrict__ s, bf16* __restrict__ d, int n) {
  int i = blockIdx.x * blockDim.x + threadIdx.x;
  int st = gridDim.x * blockDim.x;
  for (; i < n; i += st) d[i] = (bf16)s[i];
}

// ---------------------------------------------------------------------------
// Generic bf16 GEMM: C[M,N] = A[M,K] @ B[K,N] + bias, optional exact GELU.
// Stores f32 (Cf) and/or bf16 (Cb, scaled by bscale).  Tiles 128x128x32,
// 256 threads = 8 waves (2 x 4), each wave 64x32 -> 8 WMMA accumulators.
// Tile staging uses GLOBAL_LOAD_ASYNC_TO_LDS_B128 with double buffering.
// ---------------------------------------------------------------------------
#define BM 128
#define BN 128
#define BK 32

__global__ __launch_bounds__(256) void gemm_bf16_kernel(
    const bf16* __restrict__ A, const bf16* __restrict__ Bw,
    const float* __restrict__ bias, float* __restrict__ Cf,
    bf16* __restrict__ Cb, int M, int N, int K, int do_gelu, float bscale)
{
  __shared__ __align__(16) bf16 As[2][BM][BK];
  __shared__ __align__(16) bf16 Bs[2][BK][BN];

  const int tid  = threadIdx.x;
  const int lane = tid & 31;
  const int w    = tid >> 5;
  const int wm   = w & 1;     // 0..1 -> 64-row half
  const int wn   = w >> 1;    // 0..3 -> 32-col quarter
  const int m0   = blockIdx.y * BM;
  const int n0   = blockIdx.x * BN;

  v8f acc[4][2] = {};

  auto load_tile = [&](int s, int k0) {
#pragma unroll
    for (int i = 0; i < 2; ++i) {          // A: 128x32, 8 bf16 per chunk
      int chunk = tid + i * 256;
      int row = chunk >> 2, c = chunk & 3;
      async_copy_b128(&As[s][row][c * 8],
                      A + (size_t)(m0 + row) * K + k0 + c * 8);
    }
#pragma unroll
    for (int i = 0; i < 2; ++i) {          // B: 32x128
      int chunk = tid + i * 256;
      int row = chunk >> 4, c = chunk & 15;
      async_copy_b128(&Bs[s][row][c * 8],
                      Bw + (size_t)(k0 + row) * N + n0 + c * 8);
    }
  };

  load_tile(0, 0);
  async_fence();
  __syncthreads();

  const int nk = K / BK;
  for (int t = 0; t < nk; ++t) {
    const int s = t & 1;
    if (t + 1 < nk) load_tile(s ^ 1, (t + 1) * BK);   // async into other stage
    if (t + 2 < nk) {   // global_prefetch_b8 two tiles ahead
      __builtin_prefetch(A + (size_t)(m0 + (tid >> 1)) * K + (t + 2) * BK, 0, 0);
      __builtin_prefetch(Bw + (size_t)((t + 2) * BK + (tid >> 3)) * N + n0, 0, 0);
    }

    v16bf af[4];
#pragma unroll
    for (int mt = 0; mt < 4; ++mt)
      af[mt] = frag_a(&As[s][wm * 64 + mt * 16][0], BK, lane);
    v16bf bfr[2];
#pragma unroll
    for (int nt = 0; nt < 2; ++nt)
      bfr[nt] = frag_b(&Bs[s][0][wn * 32 + nt * 16], BN, lane);
#pragma unroll
    for (int mt = 0; mt < 4; ++mt)
#pragma unroll
      for (int nt = 0; nt < 2; ++nt)
        acc[mt][nt] = wmma_bf16(af[mt], bfr[nt], acc[mt][nt]);

    async_fence();     // our async stores to LDS must land before we signal
    __syncthreads();
  }

  const int hh = lane >> 4, nl = lane & 15;
#pragma unroll
  for (int mt = 0; mt < 4; ++mt) {
#pragma unroll
    for (int nt = 0; nt < 2; ++nt) {
      const int n = n0 + wn * 32 + nt * 16 + nl;
      const float bv = bias ? bias[n] : 0.0f;
#pragma unroll
      for (int r = 0; r < 8; ++r) {
        const int m = m0 + wm * 64 + mt * 16 + r + 8 * hh;
        float v = acc[mt][nt][r] + bv;
        if (do_gelu) v = 0.5f * v * (1.0f + erff(v * 0.70710678118654752f));
        const size_t idx = (size_t)m * N + n;
        if (Cf) Cf[idx] = v;
        if (Cb) Cb[idx] = (bf16)(v * bscale);
      }
    }
  }
}

// ---------------------------------------------------------------------------
// Flash-style attention.  Q is pre-scaled by 1/sqrt(DH) (folded into bf16 Q).
// Grid: (S/64, H, B).  Block: 128 threads = 4 waves; wave w owns 16 q-rows.
// Per key-tile (64 keys): S = Q K^T (WMMA), online softmax (running max/sum in
// lane registers), P restaged via LDS into A-fragment layout, O += P V (WMMA).
// K/V tiles staged with async global->LDS copies.
// ---------------------------------------------------------------------------
__global__ __launch_bounds__(128) void attn_kernel(
    const bf16* __restrict__ Q, const bf16* __restrict__ Kg,
    const bf16* __restrict__ V, const float* __restrict__ mask,
    bf16* __restrict__ ctx)
{
  __shared__ __align__(16) bf16 q_s[64][64];
  __shared__ __align__(16) bf16 k_s[64][64];     // [key][d]
  __shared__ __align__(16) bf16 v_s[64][64];     // [key][d]
  __shared__ float s_s[4][16][64];               // per-wave raw scores
  __shared__ __align__(16) bf16 p_s[4][16][64];  // per-wave probabilities
  __shared__ float alpha_s[64];
  __shared__ float linv_s[64];

  const int b = blockIdx.z, h = blockIdx.y, qt = blockIdx.x;
  const int tid = threadIdx.x, lane = tid & 31, w = tid >> 5;
  const int hc = h * D_HEAD;
  const int hh = lane >> 4, nl = lane & 15;
  const size_t qbase = (size_t)(b * S_LEN + qt * 64) * D_MODEL;

  // load Q tile (64 x 64) once
#pragma unroll
  for (int i = 0; i < 4; ++i) {
    int chunk = tid + i * 128;
    int row = chunk >> 3, c = chunk & 7;
    async_copy_b128(&q_s[row][c * 8],
                    Q + qbase + (size_t)row * D_MODEL + hc + c * 8);
  }

  v8f acc[4] = {};
  float m_run = -1e30f, l_run = 0.0f;

  for (int kt = 0; kt < S_LEN / 64; ++kt) {
    __syncthreads();   // protect k_s/v_s from previous iteration's readers
    const size_t kb = (size_t)(b * S_LEN + kt * 64) * D_MODEL;
#pragma unroll
    for (int i = 0; i < 4; ++i) {
      int chunk = tid + i * 128;
      int row = chunk >> 3, c = chunk & 7;
      async_copy_b128(&k_s[row][c * 8],
                      Kg + kb + (size_t)row * D_MODEL + hc + c * 8);
      async_copy_b128(&v_s[row][c * 8],
                      V + kb + (size_t)row * D_MODEL + hc + c * 8);
    }
    async_fence();
    __syncthreads();

    // ---- scores: 16 q-rows (this wave) x 64 keys -------------------------
    v16bf qa0 = frag_a(&q_s[w * 16][0], 64, lane);
    v16bf qa1 = frag_a(&q_s[w * 16][32], 64, lane);
    v8f sc[4];
#pragma unroll
    for (int nt = 0; nt < 4; ++nt) {
      v8f z = {};
      v16bf kb0 = frag_b_T(&k_s[nt * 16][0], 64, lane);
      v16bf kb1 = frag_b_T(&k_s[nt * 16][32], 64, lane);
      z = wmma_bf16(qa0, kb0, z);
      z = wmma_bf16(qa1, kb1, z);
      sc[nt] = z;
    }
    // scatter scores + mask to per-wave LDS
#pragma unroll
    for (int nt = 0; nt < 4; ++nt) {
      const int key = nt * 16 + nl;
      const float mv = mask[(size_t)b * S_LEN + kt * 64 + key];
#pragma unroll
      for (int r = 0; r < 8; ++r)
        s_s[w][r + 8 * hh][key] = sc[nt][r] + mv;
    }
    __syncthreads();

    // ---- online softmax: lanes 0..15 each own one q-row ------------------
    if (lane < 16) {
      const int row = lane;
      float tmax = -1e30f;
      for (int j = 0; j < 64; ++j) tmax = fmaxf(tmax, s_s[w][row][j]);
      const float mn = fmaxf(m_run, tmax);
      float sum = 0.0f;
      for (int j = 0; j < 64; ++j) {
        float p = __expf(s_s[w][row][j] - mn);
        p_s[w][row][j] = (bf16)p;
        sum += p;
      }
      const float alpha = __expf(m_run - mn);
      l_run = alpha * l_run + sum;
      m_run = mn;
      alpha_s[w * 16 + row] = alpha;
    }
    __syncthreads();

    // ---- rescale O accumulator, then O += P @ V --------------------------
    float ar[8];
#pragma unroll
    for (int r = 0; r < 8; ++r) ar[r] = alpha_s[w * 16 + r + 8 * hh];
#pragma unroll
    for (int nt = 0; nt < 4; ++nt)
#pragma unroll
      for (int r = 0; r < 8; ++r) acc[nt][r] *= ar[r];

    v16bf pa0 = frag_a(&p_s[w][0][0], 64, lane);
    v16bf pa1 = frag_a(&p_s[w][0][32], 64, lane);
#pragma unroll
    for (int nt = 0; nt < 4; ++nt) {
      v16bf vb0 = frag_b(&v_s[0][nt * 16], 64, lane);    // keys 0..31
      v16bf vb1 = frag_b(&v_s[32][nt * 16], 64, lane);   // keys 32..63
      acc[nt] = wmma_bf16(pa0, vb0, acc[nt]);
      acc[nt] = wmma_bf16(pa1, vb1, acc[nt]);
    }
  }

  if (lane < 16) linv_s[w * 16 + lane] = (l_run > 0.0f) ? (1.0f / l_run) : 0.0f;
  __syncthreads();

#pragma unroll
  for (int r = 0; r < 8; ++r) {
    const int lrow = w * 16 + r + 8 * hh;
    const float so = linv_s[lrow];
    const size_t grow = qbase + (size_t)lrow * D_MODEL + hc;
#pragma unroll
    for (int nt = 0; nt < 4; ++nt)
      ctx[grow + nt * 16 + nl] = (bf16)(acc[nt][r] * so);
  }
}

// ---------------------------------------------------------------------------
// y = LayerNorm(x + res) * g + b ; block per row of 1024, 256 threads.
// ---------------------------------------------------------------------------
__global__ __launch_bounds__(256) void add_ln_kernel(
    const float* __restrict__ x, const float* __restrict__ res,
    const float* __restrict__ g, const float* __restrict__ bta,
    float* __restrict__ outf, bf16* __restrict__ outb)
{
  __shared__ float s1[256], s2[256];
  const int row = blockIdx.x, t = threadIdx.x;
  const float* xr = x + (size_t)row * D_MODEL;
  const float* rr = res + (size_t)row * D_MODEL;

  float v[4];
  float sum = 0.0f, sq = 0.0f;
#pragma unroll
  for (int i = 0; i < 4; ++i) {
    const int c = t + i * 256;
    v[i] = xr[c] + rr[c];
    sum += v[i];
    sq  += v[i] * v[i];
  }
  s1[t] = sum; s2[t] = sq;
  __syncthreads();
  for (int o = 128; o > 0; o >>= 1) {
    if (t < o) { s1[t] += s1[t + o]; s2[t] += s2[t + o]; }
    __syncthreads();
  }
  const float mean = s1[0] * (1.0f / D_MODEL);
  const float var  = s2[0] * (1.0f / D_MODEL) - mean * mean;
  const float rstd = rsqrtf(var + 1e-12f);
#pragma unroll
  for (int i = 0; i < 4; ++i) {
    const int c = t + i * 256;
    const float y = (v[i] - mean) * rstd * g[c] + bta[c];
    if (outf) outf[(size_t)row * D_MODEL + c] = y;
    if (outb) outb[(size_t)row * D_MODEL + c] = (bf16)y;
  }
}

// ---------------------------------------------------------------------------
// Launch: full BERT layer forward.
// ---------------------------------------------------------------------------
extern "C" void kernel_launch(void* const* d_in, const int* in_sizes, int n_in,
                              void* d_out, int out_size, void* d_ws, size_t ws_size,
                              hipStream_t stream) {
  (void)in_sizes; (void)n_in; (void)out_size; (void)ws_size;

  const float* query    = (const float*)d_in[0];
  const float* key_in   = (const float*)d_in[1];
  const float* value_in = (const float*)d_in[2];
  const float* mask     = (const float*)d_in[3];
  const float* Wq = (const float*)d_in[4];  const float* bq = (const float*)d_in[5];
  const float* Wk = (const float*)d_in[6];  const float* bk = (const float*)d_in[7];
  const float* Wv = (const float*)d_in[8];  const float* bv = (const float*)d_in[9];
  const float* Wo = (const float*)d_in[10]; const float* bo = (const float*)d_in[11];
  const float* g1 = (const float*)d_in[12]; const float* b1 = (const float*)d_in[13];
  const float* Wi = (const float*)d_in[14]; const float* bi = (const float*)d_in[15];
  const float* Wd = (const float*)d_in[16]; const float* bd = (const float*)d_in[17];
  const float* g2 = (const float*)d_in[18]; const float* b2 = (const float*)d_in[19];
  float* out = (float*)d_out;

  // workspace layout (104 MB, with staged reuse)
  char* ws = (char*)d_ws;
  const size_t MB = 1ull << 20;
  bf16* xq   = (bf16*)(ws + 0);          // 8MB; reused as ctx
  bf16* xk   = (bf16*)(ws + 8 * MB);     // 8MB \ reused as attn_pre (16MB f32)
  bf16* xv   = (bf16*)(ws + 16 * MB);    // 8MB /
  bf16* wqb  = (bf16*)(ws + 24 * MB);    // 2MB each
  bf16* wkb  = (bf16*)(ws + 26 * MB);
  bf16* wvb  = (bf16*)(ws + 28 * MB);
  bf16* wob  = (bf16*)(ws + 30 * MB);
  bf16* wib  = (bf16*)(ws + 32 * MB);    // 8MB
  bf16* wdb  = (bf16*)(ws + 40 * MB);    // 8MB
  bf16* Qb   = (bf16*)(ws + 48 * MB);    // 8MB \ reused as attn_out_f (16MB f32)
  bf16* Kb   = (bf16*)(ws + 56 * MB);    // 8MB /
  bf16* Vb   = (bf16*)(ws + 64 * MB);    // 8MB; reused as attn_out_b
  bf16* inter= (bf16*)(ws + 72 * MB);    // 32MB

  bf16*  ctx        = xq;
  float* attn_pre   = (float*)(ws + 8 * MB);
  float* attn_out_f = (float*)(ws + 48 * MB);
  bf16*  attn_out_b = (bf16*)(ws + 64 * MB);
  float* layer_pre  = attn_pre;

  const int NE = M_ROWS * D_MODEL;  // 4M

  // 1) convert activations + weights to bf16
  cvt_kernel<<<1024, 256, 0, stream>>>(query,    xq,  NE);
  cvt_kernel<<<1024, 256, 0, stream>>>(key_in,   xk,  NE);
  cvt_kernel<<<1024, 256, 0, stream>>>(value_in, xv,  NE);
  cvt_kernel<<<1024, 256, 0, stream>>>(Wq, wqb, D_MODEL * D_MODEL);
  cvt_kernel<<<1024, 256, 0, stream>>>(Wk, wkb, D_MODEL * D_MODEL);
  cvt_kernel<<<1024, 256, 0, stream>>>(Wv, wvb, D_MODEL * D_MODEL);
  cvt_kernel<<<1024, 256, 0, stream>>>(Wo, wob, D_MODEL * D_MODEL);
  cvt_kernel<<<1024, 256, 0, stream>>>(Wi, wib, D_MODEL * D_FF);
  cvt_kernel<<<1024, 256, 0, stream>>>(Wd, wdb, D_FF * D_MODEL);

  // 2) QKV projections (1/sqrt(DH) folded into Q bf16 output)
  dim3 gP(D_MODEL / BN, M_ROWS / BM);
  gemm_bf16_kernel<<<gP, 256, 0, stream>>>(xq, wqb, bq, nullptr, Qb,
                                           M_ROWS, D_MODEL, D_MODEL, 0, 0.125f);
  gemm_bf16_kernel<<<gP, 256, 0, stream>>>(xk, wkb, bk, nullptr, Kb,
                                           M_ROWS, D_MODEL, D_MODEL, 0, 1.0f);
  gemm_bf16_kernel<<<gP, 256, 0, stream>>>(xv, wvb, bv, nullptr, Vb,
                                           M_ROWS, D_MODEL, D_MODEL, 0, 1.0f);

  // 3) flash attention -> ctx [B*S, H*DH]
  attn_kernel<<<dim3(S_LEN / 64, N_HEAD, BATCH), 128, 0, stream>>>(
      Qb, Kb, Vb, mask, ctx);

  // 4) output projection
  gemm_bf16_kernel<<<gP, 256, 0, stream>>>(ctx, wob, bo, attn_pre, nullptr,
                                           M_ROWS, D_MODEL, D_MODEL, 0, 1.0f);

  // 5) add & LN1 -> attn_out (f32 + bf16)
  add_ln_kernel<<<M_ROWS, 256, 0, stream>>>(attn_pre, query, g1, b1,
                                            attn_out_f, attn_out_b);

  // 6) FFN: GELU(x@Wi+bi) @ Wd + bd
  gemm_bf16_kernel<<<dim3(D_FF / BN, M_ROWS / BM), 256, 0, stream>>>(
      attn_out_b, wib, bi, nullptr, inter, M_ROWS, D_FF, D_MODEL, 1, 1.0f);
  gemm_bf16_kernel<<<gP, 256, 0, stream>>>(inter, wdb, bd, layer_pre, nullptr,
                                           M_ROWS, D_MODEL, D_FF, 0, 1.0f);

  // 7) add & LN2 -> output
  add_ln_kernel<<<M_ROWS, 256, 0, stream>>>(layer_pre, attn_out_f, g2, b2,
                                            out, nullptr);
}
